// FeatureAlign_51874615001705
// MI455X (gfx1250) — matrix operand order repaired and checked by reading, not compiled, not run on hardware
//
#include <hip/hip_runtime.h>
#include <hip/hip_bf16.h>

typedef __attribute__((ext_vector_type(16))) __bf16 v16bf;
typedef __attribute__((ext_vector_type(8)))  float  v8f;

#define B_    2
#define C_    256
#define H_    100
#define W_    100
#define HW_   10000
#define DG_   4
#define CG_   64
#define KK_   9
#define KR_G_ 576          // CG_*KK_ reduction per deform group
#define NCHG_ 18           // KR_G_/32 k-chunks per group
#define NCH_  72           // total k-chunks (2304/32)
#define MT_   16           // 256/16 M tiles
#define TILEN_ 32          // pixels per workgroup
#define NTPI_ 313          // ceil(HW_/32) tiles per image
#define EPS_  1e-5f

// ---------------- zero the group-norm sums buffer (B*32*2 floats) -----------
__global__ void k_zero(float* __restrict__ gsums) {
    if (threadIdx.x < 128) gsums[threadIdx.x] = 0.f;
}

// ---------------- offset GEMM: off[b,o,pix] = w_off[o,:] . shape[b,:,pix] ---
__global__ void k_off(const float* __restrict__ shp, const float* __restrict__ woff,
                      float* __restrict__ offb) {
    __shared__ float ws[72 * 4];
    for (int i = threadIdx.x; i < 288; i += blockDim.x) ws[i] = woff[i];
    __syncthreads();
    int idx = blockIdx.x * blockDim.x + threadIdx.x;   // b*HW + pix
    if (idx >= B_ * HW_) return;
    int b = idx / HW_, pix = idx % HW_;
    float s0 = shp[(b * 4 + 0) * HW_ + pix];
    float s1 = shp[(b * 4 + 1) * HW_ + pix];
    float s2 = shp[(b * 4 + 2) * HW_ + pix];
    float s3 = shp[(b * 4 + 3) * HW_ + pix];
    for (int o = 0; o < 72; ++o) {
        float v = ws[o*4+0]*s0 + ws[o*4+1]*s1 + ws[o*4+2]*s2 + ws[o*4+3]*s3;
        offb[((size_t)(b * 72 + o)) * HW_ + pix] = v;
    }
}

// ---- pack w_adapt (f32, [o][cf][ky][kx]) into bf16 A-fragment layout -------
// Apk[mt][kc][lane][e] per ISA 16-bit A 16x32 map:
//   m = mt*16 + (lane&15)
//   k' = (e&7) | ((lane>>4)<<3) | ((e&8)<<1)
__global__ void k_pack(const float* __restrict__ wad, __bf16* __restrict__ apk) {
    int idx = blockIdx.x * blockDim.x + threadIdx.x;
    if (idx >= MT_ * NCH_ * 32 * 16) return;
    int e    = idx & 15;
    int lane = (idx >> 4) & 31;
    int tmp  = idx >> 9;            // (mt*NCH_ + kc)
    int kc   = tmp % NCH_;
    int mt   = tmp / NCH_;
    int m  = mt * 16 + (lane & 15);
    int kq = (e & 7) | (((lane >> 4) & 1) << 3) | ((e & 8) << 1);
    int r  = kc * 32 + kq;          // global reduction index, g-major
    int g  = r / KR_G_;
    int rem = r % KR_G_;
    int kt = rem / CG_;             // tap 0..8
    int c  = rem % CG_;
    int cf = g * CG_ + c;
    apk[idx] = (__bf16)wad[((size_t)m * C_ + cf) * KK_ + kt];
}

// ---------------- fused sampling + WMMA GEMM + bias + GN partial sums -------
__global__ void __launch_bounds__(256, 1)
k_main(const float* __restrict__ x, const float* __restrict__ offb,
       const __bf16* __restrict__ apk, const float* __restrict__ badapt,
       float* __restrict__ pre, float* __restrict__ gsums) {
    // B fragments in ISA 16-bit B 32x16 layout: lane = n + 16*((k'>>3)&1), e = (k'&7) | ((k'>>4)<<3)
    __shared__ __align__(32) __bf16 ldsB[NCHG_ * 2 * 32 * 16];   // 36 KB
    __shared__ float ldsS[64];

    int wg = blockIdx.x;
    int b = wg / NTPI_;
    int pixbase = (wg % NTPI_) * TILEN_;
    int t = threadIdx.x;
    int lane = t & 31, wave = t >> 5;

    v8f acc0 = {}, acc1 = {}, acc2 = {}, acc3 = {};

    // sampler role: pixel p = t&31, channel octet cc = wave
    int p  = t & 31;
    int cc = t >> 5;
    int pix = pixbase + p;
    bool pvalid = pix < HW_;
    int h = pix / W_, w = pix % W_;

    int mt0 = wave * 2, mt1 = wave * 2 + 1;

    for (int g = 0; g < DG_; ++g) {
        __syncthreads();   // previous chunk's B fragments fully consumed
        // ---- sampling stage: fill ldsB for deform group g ----
        const float* xg = x + ((size_t)(b * C_ + g * CG_ + cc * 8)) * HW_;
        for (int k = 0; k < KK_; ++k) {
            float offy = 0.f, offx = 0.f;
            if (pvalid) {
                size_t ob = ((size_t)(b * 72 + (g * KK_ + k) * 2)) * HW_ + pix;
                offy = offb[ob];
                offx = offb[ob + HW_];
            }
            float py = (float)(h + (k / 3) - 1) + offy;
            float px = (float)(w + (k % 3) - 1) + offx;
            float y0f = floorf(py), x0f = floorf(px);
            int y0 = (int)y0f, x0 = (int)x0f;
            float wy = py - y0f, wx = px - x0f;
            float w00 = (1.f - wy) * (1.f - wx), w01 = (1.f - wy) * wx;
            float w10 = wy * (1.f - wx),         w11 = wy * wx;
            bool vy0 = (y0 >= 0) && (y0 < H_);
            bool vy1 = (y0 + 1 >= 0) && (y0 + 1 < H_);
            bool vx0 = (x0 >= 0) && (x0 < W_);
            bool vx1 = (x0 + 1 >= 0) && (x0 + 1 < W_);
            w00 = (vy0 && vx0) ? w00 : 0.f;
            w01 = (vy0 && vx1) ? w01 : 0.f;
            w10 = (vy1 && vx0) ? w10 : 0.f;
            w11 = (vy1 && vx1) ? w11 : 0.f;
            int yc0 = min(max(y0, 0), H_ - 1), yc1 = min(max(y0 + 1, 0), H_ - 1);
            int xc0 = min(max(x0, 0), W_ - 1), xc1 = min(max(x0 + 1, 0), W_ - 1);
            int i00 = yc0 * W_ + xc0, i01 = yc0 * W_ + xc1;
            int i10 = yc1 * W_ + xc0, i11 = yc1 * W_ + xc1;
            #pragma unroll
            for (int c = 0; c < 8; ++c) {
                float val = 0.f;
                if (pvalid) {
                    const float* cb = xg + (size_t)c * HW_;
                    val = w00 * cb[i00] + w01 * cb[i01] + w10 * cb[i10] + w11 * cb[i11];
                }
                int kl = k * CG_ + cc * 8 + c;     // 0..575 within group
                int chunk = kl >> 5, kq = kl & 31;
                int nt = p >> 4, n = p & 15;
                int bl = n + (((kq >> 3) & 1) << 4);
                int e  = (kq & 7) | ((kq >> 4) << 3);
                ldsB[(((chunk * 2 + nt) * 32) + bl) * 16 + e] = (__bf16)val;
            }
        }
        __syncthreads();   // B fragments ready

        // ---- GEMM: 18 k-chunks of 32, 4 WMMA tiles per wave ----
        for (int kc = 0; kc < NCHG_; ++kc) {
            int kcg = g * NCHG_ + kc;
            v16bf a0 = *(const v16bf*)(apk + (((size_t)mt0 * NCH_ + kcg) * 32 + lane) * 16);
            v16bf a1 = *(const v16bf*)(apk + (((size_t)mt1 * NCH_ + kcg) * 32 + lane) * 16);
            v16bf b0 = *(const v16bf*)(&ldsB[((kc * 2 + 0) * 32 + lane) * 16]);
            v16bf b1 = *(const v16bf*)(&ldsB[((kc * 2 + 1) * 32 + lane) * 16]);
            acc0 = __builtin_amdgcn_wmma_f32_16x16x32_bf16(false, a0, false, b0, (short)0, acc0, false, false);
            acc1 = __builtin_amdgcn_wmma_f32_16x16x32_bf16(false, a0, false, b1, (short)0, acc1, false, false);
            acc2 = __builtin_amdgcn_wmma_f32_16x16x32_bf16(false, a1, false, b0, (short)0, acc2, false, false);
            acc3 = __builtin_amdgcn_wmma_f32_16x16x32_bf16(false, a1, false, b1, (short)0, acc3, false, false);
        }
    }

    // ---- epilogue: bias, store pre-norm, group-norm partial sums ----
    __syncthreads();
    if (t < 64) ldsS[t] = 0.f;
    __syncthreads();

    auto epi = [&](v8f a, int mt, int nt) {
        int pcol = nt * 16 + (lane & 15);        // C/D layout: N = lane&15
        int pixg = pixbase + pcol;
        bool valid = pixg < HW_;
        int mb = mt * 16 + ((lane >> 4) << 3);   // lane>=16 holds rows +8
        int ng = mb >> 3;                        // all 8 rows in one GN group
        float s = 0.f, s2 = 0.f;
        #pragma unroll
        for (int r = 0; r < 8; ++r) {
            int m = mb + r;
            float v = a[r] + badapt[m];
            if (valid) {
                pre[((size_t)(b * C_ + m)) * HW_ + pixg] = v;
                s += v; s2 += v * v;
            }
        }
        atomicAdd(&ldsS[ng], s);
        atomicAdd(&ldsS[32 + ng], s2);
    };
    epi(acc0, mt0, 0); epi(acc1, mt0, 1);
    epi(acc2, mt1, 0); epi(acc3, mt1, 1);
    __syncthreads();
    if (t < 32) {
        atomicAdd(&gsums[b * 64 + t],      ldsS[t]);
        atomicAdd(&gsums[b * 64 + 32 + t], ldsS[32 + t]);
    }
}

// ---------------- finalize: group norm + affine + relu ----------------------
__global__ void k_fin(const float* __restrict__ pre, const float* __restrict__ gsums,
                      const float* __restrict__ gamma, const float* __restrict__ beta,
                      float* __restrict__ out) {
    size_t idx = (size_t)blockIdx.x * blockDim.x + threadIdx.x;
    if (idx >= (size_t)B_ * C_ * HW_) return;
    int m = (int)((idx / HW_) % C_);
    int b = (int)(idx / ((size_t)HW_ * C_));
    int ng = m >> 3;
    const float cnt = 8.f * (float)HW_;
    float mu  = gsums[b * 64 + ng] / cnt;
    float ex2 = gsums[b * 64 + 32 + ng] / cnt;
    float var = ex2 - mu * mu;
    float v = (pre[idx] - mu) * rsqrtf(var + EPS_);
    v = v * gamma[m] + beta[m];
    out[idx] = fmaxf(v, 0.f);
}

extern "C" void kernel_launch(void* const* d_in, const int* in_sizes, int n_in,
                              void* d_out, int out_size, void* d_ws, size_t ws_size,
                              hipStream_t stream) {
    const float* x     = (const float*)d_in[0];
    const float* shp   = (const float*)d_in[1];
    const float* woff  = (const float*)d_in[2];
    const float* wad   = (const float*)d_in[3];
    const float* bad   = (const float*)d_in[4];
    const float* gamma = (const float*)d_in[5];
    const float* beta  = (const float*)d_in[6];
    float* out = (float*)d_out;

    char* ws = (char*)d_ws;
    const size_t offB  = 0;                                    // 5,760,000 B
    const size_t apkB  = offB + (size_t)B_ * 72 * HW_ * 4;     // 1,179,648 B
    const size_t preB  = apkB + (size_t)MT_ * NCH_ * 32 * 16 * 2; // 20,480,000 B
    const size_t sumB  = preB + (size_t)B_ * C_ * HW_ * 4;     // 512 B

    float*  offb  = (float*)(ws + offB);
    __bf16* apk   = (__bf16*)(ws + apkB);
    float*  pre   = (float*)(ws + preB);
    float*  gsums = (float*)(ws + sumB);

    k_zero<<<1, 128, 0, stream>>>(gsums);
    k_off<<<(B_ * HW_ + 255) / 256, 256, 0, stream>>>(shp, woff, offb);
    k_pack<<<(MT_ * NCH_ * 32 * 16) / 256, 256, 0, stream>>>(wad, apk);
    k_main<<<B_ * NTPI_, 256, 0, stream>>>(x, offb, apk, bad, pre, gsums);
    k_fin<<<(B_ * C_ * HW_ + 255) / 256, 256, 0, stream>>>(pre, gsums, gamma, beta, out);
}
